// MultiHeadSelfAttention_34359738368614
// MI455X (gfx1250) — matrix operand order repaired and checked by reading, not compiled, not run on hardware
//
#include <hip/hip_runtime.h>
#include <hip/hip_bf16.h>

// ---------------- Problem constants ----------------
#define BATCH   2
#define SEQ     2048
#define DMODEL  1024
#define NHEADS  16
#define DK      64
#define BS      (BATCH * SEQ)          // 4096 rows

typedef __attribute__((ext_vector_type(16))) __bf16 v16bf;
typedef __attribute__((ext_vector_type(8)))  float  v8f;

union BF16x16 {
    v16bf v;
    uint4 q[2];
};
union BF16x8 {
    uint4 q;
    __bf16 e[8];
};
union BF16x2 {
    unsigned int u;
    __bf16 e[2];
};

// Load a 16-element bf16 fragment as two 16-byte chunks (both 16B aligned).
__device__ inline v16bf load2x8(const __bf16* p0, const __bf16* p1) {
    BF16x16 f;
    f.q[0] = *reinterpret_cast<const uint4*>(p0);
    f.q[1] = *reinterpret_cast<const uint4*>(p1);
    return f.v;
}

__device__ __forceinline__ v8f wmma_bf16(v16bf a, v16bf b, v8f c) {
    return __builtin_amdgcn_wmma_f32_16x16x32_bf16(false, a, false, b, (short)0, c, false, false);
}

// ---------------- fp32 -> bf16 cast (8 elems / thread) ----------------
__global__ void cast_f32_bf16(const float* __restrict__ in, __bf16* __restrict__ out, int n8) {
    int i = blockIdx.x * blockDim.x + threadIdx.x;
    if (i >= n8) return;
    const float4* p = reinterpret_cast<const float4*>(in) + 2 * (long)i;
    float4 a = p[0], b = p[1];
    BF16x8 o;
    o.e[0] = (__bf16)a.x; o.e[1] = (__bf16)a.y; o.e[2] = (__bf16)a.z; o.e[3] = (__bf16)a.w;
    o.e[4] = (__bf16)b.x; o.e[5] = (__bf16)b.y; o.e[6] = (__bf16)b.z; o.e[7] = (__bf16)b.w;
    reinterpret_cast<uint4*>(out)[i] = o.q;
}

// ---------------- GEMM: out[m,n] = sum_k A[m,k] * Bw[n,k]  (A @ Bw^T) ------
// Block = 128 threads (4 waves). Each wave computes a 32x64 output strip:
// 2 M-subtiles x 4 N-subtiles = 8 accumulators; per 32-wide K step we issue
// 2 A-fragment loads + 4 B-fragment loads for 8 WMMAs (0.75 loads/WMMA).
template <typename OutT>
__global__ __launch_bounds__(128)
void gemm_xwT(const __bf16* __restrict__ A, const __bf16* __restrict__ Bw,
              OutT* __restrict__ out, int M, int N, int K)
{
    const int lane = threadIdx.x & 31;
    const int wave = threadIdx.x >> 5;
    const int ln   = lane & 15;
    const int g    = lane >> 4;
    const long m0  = (long)blockIdx.x * 32;
    const long n0  = ((long)blockIdx.y * 4 + wave) * 64;

    v8f acc[2][4] = {};
    const __bf16* arow0 = A + (m0 + ln) * (long)K;
    const __bf16* arow1 = A + (m0 + 16 + ln) * (long)K;

    for (int k0 = 0; k0 < K; k0 += 32) {
        if (k0 + 64 < K) {
            __builtin_prefetch(arow0 + k0 + 64, 0, 3);   // global_prefetch_b8
            __builtin_prefetch(arow1 + k0 + 64, 0, 3);
        }
        v16bf a0 = load2x8(arow0 + k0 + g * 8, arow0 + k0 + 16 + g * 8);
        v16bf a1 = load2x8(arow1 + k0 + g * 8, arow1 + k0 + 16 + g * 8);
#pragma unroll
        for (int t = 0; t < 4; ++t) {
            const __bf16* brow = Bw + (n0 + t * 16 + ln) * (long)K + k0 + g * 8;
            v16bf b = load2x8(brow, brow + 16);
            acc[0][t] = wmma_bf16(a0, b, acc[0][t]);
            acc[1][t] = wmma_bf16(a1, b, acc[1][t]);
        }
    }

    // C layout: VGPR r, lane l -> row = r + 8*(l>=16), col = l%16
#pragma unroll
    for (int mt = 0; mt < 2; ++mt)
#pragma unroll
        for (int t = 0; t < 4; ++t)
#pragma unroll
            for (int r = 0; r < 8; ++r) {
                long row = m0 + mt * 16 + r + 8 * g;
                long col = n0 + t * 16 + ln;
                out[row * (long)N + col] = (OutT)acc[mt][t][r];
            }
}

// ---------------- RoPE + relayout [B,S,H,dk] -> [B,H,S,dk] ----------------
// oscale folds 1/sqrt(dk)*log2(e) into Q so attention scores come out of the
// QK^T WMMA already in the exp2 domain (K uses oscale = 1).
__global__ void rope_relayout(const __bf16* __restrict__ in, __bf16* __restrict__ out,
                              float oscale) {
    const int half = DK / 2;
    int idx = blockIdx.x * blockDim.x + threadIdx.x;
    int total = BATCH * SEQ * NHEADS * half;
    if (idx >= total) return;
    int i = idx % half;
    int h = (idx / half) % NHEADS;
    int s = (idx / (half * NHEADS)) % SEQ;
    int b = idx / (half * NHEADS * SEQ);

    long src = (((long)(b * SEQ + s) * NHEADS + h) * DK) + 2 * i;
    float x0 = (float)in[src];
    float x1 = (float)in[src + 1];
    float freq = __powf(10000.0f, -(float)(2 * i) / (float)DK);
    float ang  = (float)s * freq;
    float sn, cs;
    __sincosf(ang, &sn, &cs);
    long dst = (((long)(b * NHEADS + h) * SEQ + s) * DK) + 2 * i;
    out[dst]     = (__bf16)((cs * x0 - sn * x1) * oscale);
    out[dst + 1] = (__bf16)((sn * x0 + cs * x1) * oscale);
}

// ---------------- V relayout [B,S,H,dk] -> [B,H,dk,S'] (transposed) -------
// Key positions are interleaved within each 32-key block:
//   memory position a holds actual key (a&1)*16 + (a>>1),
// i.e. actual key c lands at position (c&15)*2 + ((c>>4)&1).
// This matches the packed-b32 P staging in the attention kernel; PV is
// invariant since the same permutation is applied to P's key columns.
__global__ void v_transpose(const __bf16* __restrict__ in, __bf16* __restrict__ out) {
    int idx = blockIdx.x * blockDim.x + threadIdx.x;
    int total = BATCH * SEQ * NHEADS * DK;
    if (idx >= total) return;
    int d = idx % DK;
    int h = (idx / DK) % NHEADS;
    int s = (idx / (DK * NHEADS)) % SEQ;
    int b = idx / (DK * NHEADS * SEQ);
    int sp = (s & ~31) | (((s & 15) << 1) | ((s >> 4) & 1));   // interleave in 32-block
    out[(((long)(b * NHEADS + h) * DK + d) * SEQ) + sp] = in[idx];
}

// ---------------- Flash attention step (32 keys) ----------------
// Scores arrive in the exp2 domain (scale folded into Q at RoPE time).
// Stabilizer is per 16x32 score tile (shift-invariance makes softmax exact up
// to underflow); softmax denominator accumulates via a ones-column WMMA.
// P's key columns use the interleaved enumeration: abstract position a in the
// 32-key tile = actual key (a&1)*16 + (a>>1), so each lane's (p0, p1) pair is
// stored with ONE packed ds_store_b32. V rows are pre-permuted to match.
// MASKED=true only for the diagonal tile k0 == q0.
template <bool MASKED>
__device__ __forceinline__ void attn_step(
    const __bf16* __restrict__ Kp,    // Km + bh base
    const __bf16* __restrict__ Vp,    // Vt + bh base (keys interleaved)
    __bf16* pb, int k0, int q0, int ln, int g,
    const v16bf (&aq)[2][2], const v16bf& ones,
    v8f (&acc)[2][4], v8f (&accl)[2], float (&mtile)[2])
{
    float sc[2][2][8];                 // [mt][j][r]
#pragma unroll
    for (int j = 0; j < 2; ++j) {
        const int kk = k0 + j * 16;
        const __bf16* krow = Kp + (long)(kk + ln) * DK;
        v16bf bk0 = load2x8(krow + 0  + g * 8, krow + 16 + g * 8);
        v16bf bk1 = load2x8(krow + 32 + g * 8, krow + 48 + g * 8);
#pragma unroll
        for (int mt = 0; mt < 2; ++mt) {
            v8f s = {};
            s = wmma_bf16(aq[mt][0], bk0, s);
            s = wmma_bf16(aq[mt][1], bk1, s);
#pragma unroll
            for (int r = 0; r < 8; ++r) {
                if (MASKED) {
                    int col  = kk + ln;
                    int rowq = q0 + mt * 16 + r + 8 * g;
                    sc[mt][j][r] = (col <= rowq) ? s[r] : -__builtin_inff();
                } else {
                    sc[mt][j][r] = s[r];
                }
            }
        }
    }

    unsigned int* pbu = reinterpret_cast<unsigned int*>(pb);

#pragma unroll
    for (int mt = 0; mt < 2; ++mt) {
        // per-tile stabilizer: in-lane max tree + one 5-step butterfly
        float mx = sc[mt][0][0];
#pragma unroll
        for (int j = 0; j < 2; ++j)
#pragma unroll
            for (int r = 0; r < 8; ++r) mx = fmaxf(mx, sc[mt][j][r]);
#pragma unroll
        for (int m = 1; m < 32; m <<= 1) mx = fmaxf(mx, __shfl_xor(mx, m, 32));

        // mx (and thus the branch) is uniform across the wave after reduction;
        // rescale only when the running max actually grows (fac != 1).
        if (mx > mtile[mt]) {
            float fac = __builtin_exp2f(mtile[mt] - mx);
            mtile[mt] = mx;
#pragma unroll
            for (int t = 0; t < 4; ++t)
#pragma unroll
                for (int r = 0; r < 8; ++r) acc[mt][t][r] *= fac;
#pragma unroll
            for (int r = 0; r < 8; ++r) accl[mt][r] *= fac;
        }
        const float mcur = mtile[mt];

        // exponentiate + stage P tile in LDS: interleaved key columns mean the
        // (j=0, j=1) pair for this lane is adjacent -> one b32 store per row.
#pragma unroll
        for (int r = 0; r < 8; ++r) {
            BF16x2 pk;
            pk.e[0] = (__bf16)__builtin_exp2f(sc[mt][0][r] - mcur);  // key ln
            pk.e[1] = (__bf16)__builtin_exp2f(sc[mt][1][r] - mcur);  // key 16+ln
            int prow = r + 8 * g;
            pbu[mt * 256 + prow * 16 + ln] = pk.u;
        }
    }

    // P (16x32) as A fragments from LDS
    v16bf ap0 = load2x8(pb + ln * 32 + g * 8,       pb + ln * 32 + 16 + g * 8);
    v16bf ap1 = load2x8(pb + 512 + ln * 32 + g * 8, pb + 512 + ln * 32 + 16 + g * 8);

    // softmax denominator: rowsum(P) = P x ones, via the WMMA pipe
    accl[0] = wmma_bf16(ap0, ones, accl[0]);
    accl[1] = wmma_bf16(ap1, ones, accl[1]);

    // PV: V fragments shared by both M-subtiles (keys pre-interleaved in Vt)
#pragma unroll
    for (int t = 0; t < 4; ++t) {
        const __bf16* vrow = Vp + (long)(t * 16 + ln) * SEQ + k0 + g * 8;
        v16bf bv = load2x8(vrow, vrow + 16);
        acc[0][t] = wmma_bf16(ap0, bv, acc[0][t]);
        acc[1][t] = wmma_bf16(ap1, bv, acc[1][t]);
    }
}

// ---------------- Flash attention ----------------
// Q,K: [B,H,S,DK] bf16 (RoPE'd; Q pre-scaled); Vt: [B,H,DK,S'] (interleaved);
// O: [B,S,H,DK]. Block = 256 threads (8 waves); each wave owns a 32-query tile
// and streams keys in 32-wide steps: unmasked main loop + masked diagonal step.
__global__ __launch_bounds__(256)
void flash_attn(const __bf16* __restrict__ Q, const __bf16* __restrict__ Km,
                const __bf16* __restrict__ Vt, __bf16* __restrict__ O)
{
    __shared__ alignas(16) __bf16 Pl[8][2 * 16 * 32];   // per-wave P tiles (2 x 16x32)

    const int lane = threadIdx.x & 31;
    const int wave = threadIdx.x >> 5;
    const int ln   = lane & 15;
    const int g    = lane >> 4;
    const int bh   = blockIdx.y;
    const int b    = bh >> 4;          // / NHEADS
    const int h    = bh & 15;          // % NHEADS
    const __bf16* Kp = Km + (long)bh * SEQ * DK;
    const __bf16* Vp = Vt + (long)bh * DK * SEQ;
    const int q0 = (blockIdx.x * 8 + wave) * 32;

    // Q fragments: 2 M-subtiles x 2 K-chunks covering dk=64
    v16bf aq[2][2];
#pragma unroll
    for (int mt = 0; mt < 2; ++mt) {
        const __bf16* qrow = Q + (long)bh * SEQ * DK + (long)(q0 + mt * 16 + ln) * DK;
        aq[mt][0] = load2x8(qrow + 0  + g * 8, qrow + 16 + g * 8);
        aq[mt][1] = load2x8(qrow + 32 + g * 8, qrow + 48 + g * 8);
    }

    v16bf ones;
#pragma unroll
    for (int i = 0; i < 16; ++i) ones[i] = (__bf16)1.0f;

    v8f acc[2][4] = {};
    v8f accl[2] = {};
    float mtile[2] = { -__builtin_inff(), -__builtin_inff() };

    __bf16* pb = &Pl[wave][0];

    // Unmasked main loop: every key tile strictly below the diagonal
    for (int k0 = 0; k0 < q0; k0 += 32) {
        __builtin_prefetch(Kp + (long)(k0 + 32 + ln) * DK, 0, 3);  // next K tile
        attn_step<false>(Kp, Vp, pb, k0, q0, ln, g, aq, ones, acc, accl, mtile);
    }
    // Masked diagonal tile (k0 == q0)
    attn_step<true>(Kp, Vp, pb, q0, q0, ln, g, aq, ones, acc, accl, mtile);

    // normalize + write O in [B,S,H,DK]
#pragma unroll
    for (int mt = 0; mt < 2; ++mt)
#pragma unroll
        for (int r = 0; r < 8; ++r) {
            float rinv = 1.0f / accl[mt][r];
            long srow = q0 + mt * 16 + r + 8 * g;
#pragma unroll
            for (int t = 0; t < 4; ++t) {
                long oidx = (((long)b * SEQ + srow) * NHEADS + h) * DK + t * 16 + ln;
                O[oidx] = (__bf16)(acc[mt][t][r] * rinv);
            }
        }
}

// ---------------- Host launcher ----------------
extern "C" void kernel_launch(void* const* d_in, const int* in_sizes, int n_in,
                              void* d_out, int out_size, void* d_ws, size_t ws_size,
                              hipStream_t stream) {
    const float* x  = (const float*)d_in[0];
    const float* wq = (const float*)d_in[1];
    const float* wk = (const float*)d_in[2];
    const float* wv = (const float*)d_in[3];
    const float* wo = (const float*)d_in[4];
    float* out = (float*)d_out;

    // workspace layout (bytes)
    const size_t SZ_XB  = (size_t)BS * DMODEL * 2;      // 8 MiB
    const size_t SZ_W   = (size_t)DMODEL * DMODEL * 2;  // 2 MiB
    char* ws = (char*)d_ws;
    __bf16* xb   = (__bf16*)(ws);
    __bf16* wqb  = (__bf16*)(ws + SZ_XB);
    __bf16* wkb  = (__bf16*)(ws + SZ_XB + 1 * SZ_W);
    __bf16* wvb  = (__bf16*)(ws + SZ_XB + 2 * SZ_W);
    __bf16* wob  = (__bf16*)(ws + SZ_XB + 3 * SZ_W);
    size_t off   = SZ_XB + 4 * SZ_W;
    __bf16* q0b  = (__bf16*)(ws + off);                 // [B,S,H,dk] pre-RoPE
    __bf16* k0b  = (__bf16*)(ws + off + 1 * SZ_XB);
    __bf16* v0b  = (__bf16*)(ws + off + 2 * SZ_XB);
    __bf16* qr   = (__bf16*)(ws + off + 3 * SZ_XB);     // [B,H,S,dk] RoPE'd
    __bf16* kr   = (__bf16*)(ws + off + 4 * SZ_XB);
    __bf16* vt   = q0b;   // alias: q0b dead after RoPE(Q)
    __bf16* attn = k0b;   // alias: k0b dead after RoPE(K)

    const int nx = BS * DMODEL;        // 4,194,304
    const int nw = DMODEL * DMODEL;    // 1,048,576

    // 1) casts (8 elements per thread)
    cast_f32_bf16<<<(nx / 8 + 255) / 256, 256, 0, stream>>>(x,  xb,  nx / 8);
    cast_f32_bf16<<<(nw / 8 + 255) / 256, 256, 0, stream>>>(wq, wqb, nw / 8);
    cast_f32_bf16<<<(nw / 8 + 255) / 256, 256, 0, stream>>>(wk, wkb, nw / 8);
    cast_f32_bf16<<<(nw / 8 + 255) / 256, 256, 0, stream>>>(wv, wvb, nw / 8);
    cast_f32_bf16<<<(nw / 8 + 255) / 256, 256, 0, stream>>>(wo, wob, nw / 8);

    // 2) QKV projections: y = x @ W^T   (M=4096, N=1024, K=1024)
    dim3 ggrid(BS / 32, DMODEL / 256);
    gemm_xwT<__bf16><<<ggrid, 128, 0, stream>>>(xb, wqb, q0b, BS, DMODEL, DMODEL);
    gemm_xwT<__bf16><<<ggrid, 128, 0, stream>>>(xb, wkb, k0b, BS, DMODEL, DMODEL);
    gemm_xwT<__bf16><<<ggrid, 128, 0, stream>>>(xb, wvb, v0b, BS, DMODEL, DMODEL);

    // 3) RoPE + relayout; Q gets 1/sqrt(dk) * log2(e) folded in
    const float QSCALE = 0.125f * 1.44269504088896f;
    const int npairs = BATCH * SEQ * NHEADS * (DK / 2);
    rope_relayout<<<(npairs + 255) / 256, 256, 0, stream>>>(q0b, qr, QSCALE);
    rope_relayout<<<(npairs + 255) / 256, 256, 0, stream>>>(k0b, kr, 1.0f);

    // 4) V -> [B,H,dk,S'] with per-32-block key interleave
    v_transpose<<<(nx + 255) / 256, 256, 0, stream>>>(v0b, vt);

    // 5) flash attention (32 query rows per wave, 256 per block)
    dim3 agrid(SEQ / 256, BATCH * NHEADS);
    flash_attn<<<agrid, 256, 0, stream>>>(qr, kr, vt, attn);

    // 6) output projection -> fp32 d_out
    gemm_xwT<float><<<ggrid, 128, 0, stream>>>(attn, wob, out, BS, DMODEL, DMODEL);

    (void)in_sizes; (void)n_in; (void)out_size; (void)ws_size;
}
